// GNNWithTemporalAttention_75788992905534
// MI455X (gfx1250) — matrix operand order repaired and checked by reading, not compiled, not run on hardware
//
#include <hip/hip_runtime.h>
#include <hip/hip_bf16.h>
#include <math.h>

// ---------------- problem constants (from reference) ----------------
#define NN      51200          // nodes
#define EE      409600         // edges
#define INC     512
#define HID     256
#define OUTC    256
#define HEADS   2
#define NAUD    64
#define NPA     800            // nodes per audio
#define NEG_SLOPE 0.2f

// ---- GEMM staging geometry ----
#define KP       64                       // K-panel (bf16 halves)
#define BROWS    512                      // B rows staged (full Nout)
#define BPAD     8                        // pad halves/row (16B) -> conflict-free banks
#define BSTRIDE  (KP + BPAD)              // 72 halves = 144 B row stride
#define BUFELTS  (BROWS * BSTRIDE)        // halves per buffer
#define BUFBYTES (BUFELTS * 2)            // 73728 B
#define LDSBYTES (2 * BUFBYTES)           // 147456 B (double buffered)

typedef __attribute__((ext_vector_type(16))) __bf16 v16bf;
typedef __attribute__((ext_vector_type(8)))  __bf16 v8bf;
typedef __attribute__((ext_vector_type(8)))  float  v8f;

// ---------------- helpers ----------------
__device__ __forceinline__ __bf16 f2bf(float f) {
    unsigned u = __float_as_uint(f);
    unsigned r = (u + 0x7FFFu + ((u >> 16) & 1u)) >> 16;   // RNE
    unsigned short us = (unsigned short)r;
    __bf16 b;
    __builtin_memcpy(&b, &us, 2);
    return b;
}
__device__ __forceinline__ float lrelu(float x) { return x > 0.f ? x : NEG_SLOPE * x; }
// order-preserving float -> uint key (for atomic max on floats of any sign)
__device__ __forceinline__ unsigned fkey(float f) {
    unsigned u = __float_as_uint(f);
    return (u & 0x80000000u) ? ~u : (u | 0x80000000u);
}
__device__ __forceinline__ float kinv(unsigned k) {
    unsigned u = (k & 0x80000000u) ? (k & 0x7FFFFFFFu) : ~k;
    return __uint_as_float(u);
}

// A-operand fragment (16x32 bf16), ISA 7.12.2: lane L holds row M=L%16,
// elements 0..7 = K[g*8 .. g*8+7], elements 8..15 = K[16+g*8 .. 16+g*8+7], g=L/16.
__device__ __forceinline__ v16bf load_fragA(const __bf16* __restrict__ row, int kbase) {
    v8bf lo = *reinterpret_cast<const v8bf*>(row + kbase);
    v8bf hi = *reinterpret_cast<const v8bf*>(row + kbase + 16);
    v16bf f;
#pragma unroll
    for (int i = 0; i < 8; ++i) { f[i] = lo[i]; f[i + 8] = hi[i]; }
    return f;
}

// ---------------- conversion kernels ----------------
__global__ void k_cvt_bf16(const float* __restrict__ src, __bf16* __restrict__ dst, int n) {
    int i = blockIdx.x * 256 + threadIdx.x;
    if (i < n) dst[i] = f2bf(src[i]);
}

// W[K][Nout] (row-major, f32) -> Wt[Nout][K] (bf16)
__global__ void k_transpose_cvt(const float* __restrict__ W, __bf16* __restrict__ Wt,
                                int K, int Nout) {
    int i = blockIdx.x * 256 + threadIdx.x;
    if (i >= K * Nout) return;
    int n = i / K, k = i - n * K;
    Wt[i] = f2bf(W[(size_t)k * Nout + n]);
}

// ---------------- WMMA GEMM: C[M][512] = A[M][K] * Bt[512][K]^T ----------------
// block = 256 threads (8 waves): 2 waves (M) x 4 (N) -> 32 x 512 tile.
// B panels are double-buffered in LDS via global_load_async_to_lds_b128 (ASYNCcnt);
// WMMAs consume B from LDS (ds_load_b128), A streams from HBM once.
__global__ void k_gemm_bf16(const __bf16* __restrict__ A, const __bf16* __restrict__ Bt,
                            float* __restrict__ C, int M, int K) {
    extern __shared__ __bf16 smem[];           // 2 * BUFELTS halves
    const int Nout = 512;
    const int tid  = threadIdx.x;
    const int lane = tid & 31;
    const int wave = tid >> 5;
    const int wm = wave >> 2;                  // 0..1
    const int wn = wave & 3;                   // 0..3
    const int m0 = blockIdx.x * 32 + wm * 16;
    const int n0 = wn * 128;
    const int l15 = lane & 15;
    const int g8  = (lane >> 4) * 8;
    const int g16 = (lane >> 4) * 16;

    const __bf16* arow = A + (size_t)(m0 + l15) * K;
    const unsigned lds0 = (unsigned)(size_t)smem;

    const int P = K / KP;                      // panels (8 or 4)

    // async-copy one K-panel of Bt into LDS buffer b (16 instr/wave, 4096 x 16B total)
    auto issue_panel = [&](int p, int b) {
        const __bf16* gbase = Bt + (size_t)p * KP;
        unsigned lbase = lds0 + (unsigned)(b * BUFBYTES);
#pragma unroll
        for (int i = 0; i < 16; ++i) {
            int linear = tid + i * 256;                       // 0..4095
            int row = linear >> 3;                            // 8 segs/row
            int seg = linear & 7;
            const __bf16* g = gbase + (size_t)row * K + seg * 8;
            unsigned l = lbase + (unsigned)(row * (BSTRIDE * 2) + seg * 16);
            asm volatile("global_load_async_to_lds_b128 %0, %1, off"
                         :: "v"(l), "v"(g) : "memory");
        }
    };

    issue_panel(0, 0);
    if (P > 1) issue_panel(1, 1);

    v8f c[8];
#pragma unroll
    for (int j = 0; j < 8; ++j) c[j] = (v8f){};

    for (int p = 0; p < P; ++p) {
        if (p + 1 < P) asm volatile("s_wait_asynccnt 0x10" ::: "memory");  // panel p done
        else           asm volatile("s_wait_asynccnt 0x0"  ::: "memory");
        __syncthreads();                                     // all waves' copies visible
        const __bf16* buf = smem + (size_t)(p & 1) * BUFELTS;
#pragma unroll
        for (int s = 0; s < 2; ++s) {                        // two K=32 steps per panel
            const int k = p * KP + s * 32;
            if (k + 96 < K) __builtin_prefetch(arow + k + 96, 0, 3);
            v16bf a = load_fragA(arow, k + g8);
#pragma unroll
            for (int j = 0; j < 8; ++j) {
                const __bf16* brow = buf + (n0 + j * 16 + l15) * BSTRIDE + s * 32 + g8;
                v8bf lo = *reinterpret_cast<const v8bf*>(brow);
                v8bf hi = *reinterpret_cast<const v8bf*>(brow + 16);
                v16bf bb;
#pragma unroll
                for (int i = 0; i < 8; ++i) { bb[i] = lo[i]; bb[i + 8] = hi[i]; }
                c[j] = __builtin_amdgcn_wmma_f32_16x16x32_bf16(false, a, false, bb,
                                                               (short)0, c[j], false, false);
            }
        }
        __syncthreads();                                     // done reading buf before reuse
        if (p + 2 < P) issue_panel(p + 2, p & 1);
    }

    // C/D layout: lanes 0..15 -> M=r, lanes 16..31 -> M=8+r; N = lane%16
    const int crow = m0 + (lane >> 4) * 8;
    const int ccol = n0 + l15;
#pragma unroll
    for (int r = 0; r < 8; ++r) {
        float* cp = C + (size_t)(crow + r) * Nout + ccol;
#pragma unroll
        for (int j = 0; j < 8; ++j) cp[j * 16] = c[j][r];
    }
}

// ---------------- attention scores: als/ald[n][h] = sum_c h[n][h*256+c]*a_{s,d}[h][c] ----
__global__ void k_attn_scores(const float* __restrict__ h, const float* __restrict__ a_s,
                              const float* __restrict__ a_d, float* __restrict__ als,
                              float* __restrict__ ald) {
    int wave = threadIdx.x >> 5, lane = threadIdx.x & 31;
    int n = blockIdx.x * 8 + wave;
    if (n >= NN) return;
    const float* hn = h + (size_t)n * (HEADS * HID);
    float s0 = 0, s1 = 0, d0 = 0, d1 = 0;
    for (int i = lane; i < HID; i += 32) {
        float v0 = hn[i], v1 = hn[HID + i];
        s0 += v0 * a_s[i];        s1 += v1 * a_s[HID + i];
        d0 += v0 * a_d[i];        d1 += v1 * a_d[HID + i];
    }
#pragma unroll
    for (int m = 16; m; m >>= 1) {
        s0 += __shfl_xor(s0, m); s1 += __shfl_xor(s1, m);
        d0 += __shfl_xor(d0, m); d1 += __shfl_xor(d1, m);
    }
    if (!lane) {
        als[n * 2] = s0; als[n * 2 + 1] = s1;
        ald[n * 2] = d0; ald[n * 2 + 1] = d1;
    }
}

// ---------------- softmax over incoming edges (with self loops) ----------------
__global__ void k_self_init_max(const float* __restrict__ als, const float* __restrict__ ald,
                                unsigned* __restrict__ mkey) {
    int n = blockIdx.x * 256 + threadIdx.x;
    if (n >= NN) return;
    mkey[n * 2]     = fkey(lrelu(als[n * 2]     + ald[n * 2]));
    mkey[n * 2 + 1] = fkey(lrelu(als[n * 2 + 1] + ald[n * 2 + 1]));
}

__global__ void k_edge_max(const int* __restrict__ src, const int* __restrict__ dst,
                           const float* __restrict__ als, const float* __restrict__ ald,
                           unsigned* __restrict__ mkey) {
    int e = blockIdx.x * 256 + threadIdx.x;
    if (e >= EE) return;
    int s = src[e], d = dst[e];
    atomicMax(&mkey[d * 2],     fkey(lrelu(als[s * 2]     + ald[d * 2])));
    atomicMax(&mkey[d * 2 + 1], fkey(lrelu(als[s * 2 + 1] + ald[d * 2 + 1])));
}

__global__ void k_self_exp(const float* __restrict__ als, const float* __restrict__ ald,
                           const unsigned* __restrict__ mkey, float* __restrict__ wself,
                           float* __restrict__ den) {
    int n = blockIdx.x * 256 + threadIdx.x;
    if (n >= NN) return;
#pragma unroll
    for (int h = 0; h < 2; ++h) {
        float e = lrelu(als[n * 2 + h] + ald[n * 2 + h]);
        float w = expf(e - kinv(mkey[n * 2 + h]));
        wself[n * 2 + h] = w;
        den[n * 2 + h]   = w;           // init denominator with self-loop term
    }
}

__global__ void k_edge_exp(const int* __restrict__ src, const int* __restrict__ dst,
                           const float* __restrict__ als, const float* __restrict__ ald,
                           const unsigned* __restrict__ mkey, float* __restrict__ wE,
                           float* __restrict__ den) {
    int e = blockIdx.x * 256 + threadIdx.x;
    if (e >= EE) return;
    int s = src[e], d = dst[e];
#pragma unroll
    for (int h = 0; h < 2; ++h) {
        float v = lrelu(als[s * 2 + h] + ald[d * 2 + h]);
        float w = expf(v - kinv(mkey[d * 2 + h]));
        wE[e * 2 + h] = w;
        atomicAdd(&den[d * 2 + h], w);
    }
}

// agg[n][ch] = wself[n][ch/256] * h[n][ch]   (self-loop contribution, also clears agg)
__global__ void k_agg_init(const float* __restrict__ h, const float* __restrict__ wself,
                           float* __restrict__ agg) {
    int i = blockIdx.x * 256 + threadIdx.x;
    if (i >= NN * 512) return;
    int n = i >> 9, ch = i & 511;
    agg[i] = wself[n * 2 + (ch >> 8)] * h[i];
}

// one block (256 threads) per edge: agg[dst] += w * h[src]
__global__ void k_edge_agg(const int* __restrict__ src, const int* __restrict__ dst,
                           const float* __restrict__ wE, const float* __restrict__ h,
                           float* __restrict__ agg) {
    int e = blockIdx.x;
    int s = src[e], d = dst[e];
    float w0 = wE[e * 2], w1 = wE[e * 2 + 1];
    const float* hs = h + (size_t)s * 512;
    float* ad = agg + (size_t)d * 512;
    int t = threadIdx.x;
    atomicAdd(&ad[t],       w0 * hs[t]);
    atomicAdd(&ad[t + 256], w1 * hs[t + 256]);
}

// head-mean + bias (+ optional relu); optionally write f32 and/or bf16 copies
__global__ void k_finalize(const float* __restrict__ agg, const float* __restrict__ den,
                           const float* __restrict__ bias, float* __restrict__ out_f32,
                           __bf16* __restrict__ out_bf, int relu) {
    int i = blockIdx.x * 256 + threadIdx.x;
    if (i >= NN * 256) return;
    int n = i >> 8, c = i & 255;
    float v = 0.5f * (agg[(size_t)n * 512 + c]       / den[n * 2] +
                      agg[(size_t)n * 512 + 256 + c] / den[n * 2 + 1]) + bias[c];
    if (relu) v = v > 0.f ? v : 0.f;
    if (out_f32) out_f32[i] = v;
    if (out_bf)  out_bf[i]  = f2bf(v);
}

// ---------------- temporal attention ----------------
__global__ void k_logits(const float* __restrict__ emb, const float* __restrict__ w_att,
                         const float* __restrict__ b_att, float* __restrict__ logits) {
    int wave = threadIdx.x >> 5, lane = threadIdx.x & 31;
    int n = blockIdx.x * 8 + wave;
    if (n >= NN) return;
    const float* en = emb + (size_t)n * OUTC;
    float s = 0.f;
    for (int i = lane; i < OUTC; i += 32) s += en[i] * w_att[i];
#pragma unroll
    for (int m = 16; m; m >>= 1) s += __shfl_xor(s, m);
    if (!lane) logits[n] = s + b_att[0];
}

// one block per audio: softmax over its 800 nodes, attend, classify
__global__ void k_audio(const float* __restrict__ emb, const float* __restrict__ logits,
                        const float* __restrict__ w_cls, const float* __restrict__ b_cls,
                        float* __restrict__ out_audio) {
    __shared__ float sw[NPA];
    __shared__ float red[256];
    int a = blockIdx.x, t = threadIdx.x;
    int base = a * NPA;

    float m = -INFINITY;
    for (int i = t; i < NPA; i += 256) m = fmaxf(m, logits[base + i]);
    red[t] = m; __syncthreads();
    for (int s = 128; s; s >>= 1) { if (t < s) red[t] = fmaxf(red[t], red[t + s]); __syncthreads(); }
    m = red[0]; __syncthreads();

    float ssum = 0.f;
    for (int i = t; i < NPA; i += 256) { float w = expf(logits[base + i] - m); sw[i] = w; ssum += w; }
    red[t] = ssum; __syncthreads();
    for (int s = 128; s; s >>= 1) { if (t < s) red[t] += red[t + s]; __syncthreads(); }
    float den = red[0]; __syncthreads();

    float acc = 0.f;                     // attended[c = t]
    for (int i = 0; i < NPA; ++i) acc += sw[i] * emb[(size_t)(base + i) * OUTC + t];
    acc /= den;

    red[t] = acc * w_cls[t * 2]; __syncthreads();
    for (int s = 128; s; s >>= 1) { if (t < s) red[t] += red[t + s]; __syncthreads(); }
    float r0 = red[0]; __syncthreads();
    red[t] = acc * w_cls[t * 2 + 1]; __syncthreads();
    for (int s = 128; s; s >>= 1) { if (t < s) red[t] += red[t + s]; __syncthreads(); }
    if (t == 0) {
        out_audio[a * 2]     = r0     + b_cls[0];
        out_audio[a * 2 + 1] = red[0] + b_cls[1];
    }
}

// ---------------- launcher ----------------
extern "C" void kernel_launch(void* const* d_in, const int* in_sizes, int n_in,
                              void* d_out, int out_size, void* d_ws, size_t ws_size,
                              hipStream_t stream) {
    const float* x     = (const float*)d_in[0];
    const int*   eidx  = (const int*)  d_in[1];
    const float* W1    = (const float*)d_in[3];
    const float* a_s1  = (const float*)d_in[4];
    const float* a_d1  = (const float*)d_in[5];
    const float* b1    = (const float*)d_in[6];
    const float* W2    = (const float*)d_in[7];
    const float* a_s2  = (const float*)d_in[8];
    const float* a_d2  = (const float*)d_in[9];
    const float* b2    = (const float*)d_in[10];
    const float* w_att = (const float*)d_in[11];
    const float* b_att = (const float*)d_in[12];
    const float* w_cls = (const float*)d_in[13];
    const float* b_cls = (const float*)d_in[14];

    const int* src = eidx;
    const int* dst = eidx + EE;

    float* node_emb  = (float*)d_out;             // [NN, 256]
    float* out_audio = node_emb + (size_t)NN * OUTC;

    // workspace carve-out (256B aligned)
    char*  w   = (char*)d_ws;
    size_t off = 0;
    auto take = [&](size_t bytes) -> void* {
        void* p = w + off;
        off += (bytes + 255) & ~(size_t)255;
        return p;
    };
    __bf16*   xb    = (__bf16*)  take((size_t)NN * INC * 2);          // bf16 x
    __bf16*   w1t   = (__bf16*)  take((size_t)INC * 512 * 2);         // W1^T [512][512]
    __bf16*   w2t   = (__bf16*)  take((size_t)HID * 512 * 2);         // W2^T [512][256]
    __bf16*   hb    = (__bf16*)  take((size_t)NN * HID * 2);          // bf16 layer1 output
    float*    hbuf  = (float*)   take((size_t)NN * 512 * 4);          // h (pre-agg features)
    float*    agg   = (float*)   take((size_t)NN * 512 * 4);          // scatter accumulator
    float*    als   = (float*)   take((size_t)NN * 2 * 4);
    float*    ald   = (float*)   take((size_t)NN * 2 * 4);
    unsigned* mkey  = (unsigned*)take((size_t)NN * 2 * 4);
    float*    den   = (float*)   take((size_t)NN * 2 * 4);
    float*    wself = (float*)   take((size_t)NN * 2 * 4);
    float*    wE    = (float*)   take((size_t)EE * 2 * 4);
    float*    logit = (float*)   take((size_t)NN * 4);
    (void)ws_size; (void)in_sizes; (void)n_in; (void)out_size;

    const dim3 B(256);

    // allow >64KB dynamic LDS for the staged GEMM (not a stream op; capture-safe)
    static int lds_attr_set = 0;
    (void)lds_attr_set;
    hipFuncSetAttribute((const void*)k_gemm_bf16,
                        hipFuncAttributeMaxDynamicSharedMemorySize, LDSBYTES);

    // ---- precision conversion ----
    k_cvt_bf16<<<(NN * INC + 255) / 256, B, 0, stream>>>(x, xb, NN * INC);
    k_transpose_cvt<<<(INC * 512 + 255) / 256, B, 0, stream>>>(W1, w1t, INC, 512);
    k_transpose_cvt<<<(HID * 512 + 255) / 256, B, 0, stream>>>(W2, w2t, HID, 512);

    // ================= layer 1 =================
    k_gemm_bf16<<<NN / 32, B, LDSBYTES, stream>>>(xb, w1t, hbuf, NN, INC);
    k_attn_scores<<<NN / 8, B, 0, stream>>>(hbuf, a_s1, a_d1, als, ald);
    k_self_init_max<<<(NN + 255) / 256, B, 0, stream>>>(als, ald, mkey);
    k_edge_max<<<(EE + 255) / 256, B, 0, stream>>>(src, dst, als, ald, mkey);
    k_self_exp<<<(NN + 255) / 256, B, 0, stream>>>(als, ald, mkey, wself, den);
    k_edge_exp<<<(EE + 255) / 256, B, 0, stream>>>(src, dst, als, ald, mkey, wE, den);
    k_agg_init<<<(NN * 512 + 255) / 256, B, 0, stream>>>(hbuf, wself, agg);
    k_edge_agg<<<EE, B, 0, stream>>>(src, dst, wE, hbuf, agg);
    k_finalize<<<(NN * 256 + 255) / 256, B, 0, stream>>>(agg, den, b1, nullptr, hb, 1);

    // ================= layer 2 =================
    k_gemm_bf16<<<NN / 32, B, LDSBYTES, stream>>>(hb, w2t, hbuf, NN, HID);
    k_attn_scores<<<NN / 8, B, 0, stream>>>(hbuf, a_s2, a_d2, als, ald);
    k_self_init_max<<<(NN + 255) / 256, B, 0, stream>>>(als, ald, mkey);
    k_edge_max<<<(EE + 255) / 256, B, 0, stream>>>(src, dst, als, ald, mkey);
    k_self_exp<<<(NN + 255) / 256, B, 0, stream>>>(als, ald, mkey, wself, den);
    k_edge_exp<<<(EE + 255) / 256, B, 0, stream>>>(src, dst, als, ald, mkey, wE, den);
    k_agg_init<<<(NN * 512 + 255) / 256, B, 0, stream>>>(hbuf, wself, agg);
    k_edge_agg<<<EE, B, 0, stream>>>(src, dst, wE, hbuf, agg);
    k_finalize<<<(NN * 256 + 255) / 256, B, 0, stream>>>(agg, den, b2, node_emb, nullptr, 0);

    // ================= temporal attention + classifier =================
    k_logits<<<NN / 8, B, 0, stream>>>(node_emb, w_att, b_att, logit);
    k_audio<<<NAUD, B, 0, stream>>>(node_emb, logit, w_cls, b_cls, out_audio);
}
